// GRUFeatureExtractor_88338887344757
// MI455X (gfx1250) — matrix operand order repaired and checked by reading, not compiled, not run on hardware
//
#include <hip/hip_runtime.h>
#include <hip/hip_bf16.h>

typedef _Float16 v16h __attribute__((ext_vector_type(16)));
typedef _Float16 h8   __attribute__((ext_vector_type(8)));
typedef _Float16 h2   __attribute__((ext_vector_type(2)));
typedef float    v8f  __attribute__((ext_vector_type(8)));

#define T_LEN 30
#define HID 5
#define NGATE 15
#define WAVES 2                            // waves per block
#define REC 16                             // halves per (t,batch) record (10 used + zero pad)
#define SEQ_HALVES (T_LEN * 16 * REC)      // 7680 halves per buffer per wave
#define WAVE_HALVES (2 * SEQ_HALVES + REC) // 2 buffers + 1 all-zero record

__device__ __forceinline__ float fastrcp(float x) {
#if __has_builtin(__builtin_amdgcn_rcpf)
  return __builtin_amdgcn_rcpf(x);
#else
  return 1.0f / x;
#endif
}
__device__ __forceinline__ float sigf(float x) {
  return fastrcp(1.0f + __expf(-x));       // v_exp_f32 + v_rcp_f32
}
__device__ __forceinline__ float tanh_fast(float x) {
#if __has_builtin(__builtin_amdgcn_tanhf)
  return __builtin_amdgcn_tanhf(x);        // gfx1250 v_tanh_f32 (confirmed in disasm)
#else
  return 1.0f - 2.0f * fastrcp(__expf(2.0f * x) + 1.0f);
#endif
}

__global__ __launch_bounds__(WAVES * 32)
void gru_bidir_fused(const float* __restrict__ x,
                     const float* __restrict__ w_ih0, const float* __restrict__ w_hh0,
                     const float* __restrict__ b_ih0, const float* __restrict__ b_hh0,
                     const float* __restrict__ w_ihL, const float* __restrict__ w_hhL,
                     const float* __restrict__ b_ihL, const float* __restrict__ b_hhL,
                     float* __restrict__ out)
{
  __shared__ _Float16 lds[WAVES * WAVE_HALVES];

  const int tid   = threadIdx.x;
  const int wave  = tid >> 5;
  const int lane  = tid & 31;
  const int bat   = lane & 15;   // batch-in-wave (WMMA N); lanes b and b+16 share it
  const bool hi   = lane >= 16;  // hi half-wave: D rows 8..15, B K=16..31 (all zero)
  const int gate  = bat;         // A-matrix row (M) owned by this lane

  // All LDS addressing is integer offsets into `lds` so every access stays in
  // addrspace(3) (ds_load/ds_store); loop-carried pointers regressed to FLAT.
  const int wbase = wave * WAVE_HALVES;
  const int ZOFF  = wbase + 2 * SEQ_HALVES; // 16-half record that stays zero forever
  int inBase  = wbase;
  int outBase = wbase + SEQ_HALVES;

  // Zero the per-wave region (records' pad halves must be 0; zrec must be 0).
  {
    unsigned int* p = (unsigned int*)&lds[wbase];
    for (int i = lane; i < WAVE_HALVES / 2; i += 32) p[i] = 0u;
  }

  // Stage layer-0 input: x is [B,1,T]; record (t,b) gets x at k=0, k=1..15 stay 0.
  const long gbat0 = (long)(blockIdx.x * WAVES + wave) * 16;
  for (int idx = lane; idx < 16 * T_LEN; idx += 32) {
    int b = idx / T_LEN, t = idx % T_LEN;
    lds[inBase + (t * 16 + b) * REC] = (_Float16)x[(gbat0 + b) * T_LEN + t];
  }

  #pragma unroll 1
  for (int layer = 0; layer < 5; ++layer) {
    const int D = (layer == 0) ? 1 : 10;   // input feature count this layer

    // ---- Per-direction operand setup (registers, indexed by unrolled dir) ----
    // B carries x_t at K=0..9 and h at K=10..14 (single shared B operand).
    // A1: rows 0..9  = [W_ih_rz | W_hh_rz]   (full r/z preact in one WMMA)
    //     rows 10..14= [W_ih_n  | 0 ]        (i_n only)
    // A2: rows 0..4  = [0 | W_hn]            (h_n lands on LOW lane c[0..4])
    v16h a1[2], a2[2];
    v8f  c1[2], c2[2];
    #pragma unroll
    for (int dir = 0; dir < 2; ++dir) {
      const int li = (layer - 1) * 2 + dir;
      const float* wih = (layer == 0) ? (w_ih0 + dir * NGATE * 1)   : (w_ihL + li * NGATE * 10);
      const float* whh = (layer == 0) ? (w_hh0 + dir * NGATE * HID) : (w_hhL + li * NGATE * HID);
      const float* bih = (layer == 0) ? (b_ih0 + dir * NGATE)       : (b_ihL + li * NGATE);
      const float* bhh = (layer == 0) ? (b_hh0 + dir * NGATE)       : (b_hhL + li * NGATE);

      v16h t1 = {}, t2 = {};
      if (!hi) {
        #pragma unroll
        for (int i = 0; i < 8; ++i)
          if (gate < NGATE && i < D) t1[i] = (_Float16)wih[gate * D + i];
        if (gate < NGATE && D > 8) {       // K=8,9 -> input features 8,9 (all 15 rows)
          t1[8] = (_Float16)wih[gate * D + 8];
          t1[9] = (_Float16)wih[gate * D + 9];
        }
        if (gate < 10) {                   // K=10..14 -> W_hh for the r/z rows
          #pragma unroll
          for (int i = 10; i <= 14; ++i) t1[i] = (_Float16)whh[gate * HID + (i - 10)];
        }
        if (gate < HID) {                  // A2 row g: W_hh row (10+g) at K=10..14
          #pragma unroll
          for (int i = 10; i <= 14; ++i) t2[i] = (_Float16)whh[(10 + gate) * HID + (i - 10)];
        }
      }
      a1[dir] = t1;
      a2[dir] = t2;
      #pragma unroll
      for (int r = 0; r < 8; ++r) {
        int g = r + (hi ? 8 : 0);
        c1[dir][r] = (g < 10) ? (bih[g] + bhh[g]) : ((g < NGATE) ? bih[g] : 0.0f);
        c2[dir][r] = (!hi && r < HID) ? bhh[10 + r] : 0.0f;
      }
    }

    // ---- Both directions advance together: 2 independent chains -> 2x ILP ----
    float h[2][HID];
    #pragma unroll
    for (int d = 0; d < 2; ++d)
      #pragma unroll
      for (int j = 0; j < HID; ++j) h[d][j] = 0.0f;

    // Integer offsets (halves) carried through the loop; pointers rebuilt from
    // `lds` each iteration so the accesses stay DS.
    int ro[2], wo[2], rs[2], ws[2];
    ro[0] = hi ? ZOFF : (inBase + bat * REC);                        // t = 0, fwd
    ro[1] = hi ? ZOFF : (inBase + ((T_LEN - 1) * 16 + bat) * REC);   // t = T-1, bwd
    wo[0] = outBase + bat * REC;                                     // fwd -> halves 0..4
    wo[1] = outBase + ((T_LEN - 1) * 16 + bat) * REC + HID;          // bwd -> halves 5..9
    rs[0] = hi ? 0 : (16 * REC);
    rs[1] = hi ? 0 : -(16 * REC);
    ws[0] = 16 * REC;
    ws[1] = -(16 * REC);

    for (int step = 0; step < T_LEN; ++step) {
      #pragma unroll
      for (int d = 0; d < 2; ++d) {
        // B: K=0..9 from the record (hi lanes read the zero record -> K=16..31==0),
        //    K=10..14 = current hidden state, K=15 = 0.
        h8 r0 = *(const h8*)(&lds[ro[d]]);
        h2 r1 = *(const h2*)(&lds[ro[d] + 8]);
        v16h b;
        #pragma unroll
        for (int i = 0; i < 8; ++i) b[i] = r0[i];
        b[8] = r1[0];
        b[9] = r1[1];
        #pragma unroll
        for (int j = 0; j < HID; ++j) b[10 + j] = hi ? (_Float16)0 : (_Float16)h[d][j];
        b[15] = (_Float16)0;

        // d1: low c[0..4]=r preact, c[5..7]=z0..2; hi c[0..1]=z3,z4 preact, c[2..6]=i_n
        v8f d1 = __builtin_amdgcn_wmma_f32_16x16x32_f16(
            false, a1[d], false, b, (short)0, c1[d], false, false);
        // d2: low c[0..4] = h_n preact (incl. b_hn)
        v8f d2 = __builtin_amdgcn_wmma_f32_16x16x32_f16(
            false, a2[d], false, b, (short)0, c2[d], false, false);

        float sg[8];
        #pragma unroll
        for (int r = 0; r < 8; ++r) sg[r] = sigf(d1[r]);

        // One exchange round, hi -> low: i_n (5) + z3 + z4. Uniform send, no masks.
        float oth[7];
        #pragma unroll
        for (int j = 0; j < 7; ++j) {
          float send = (j < 5) ? d1[2 + j] : ((j == 5) ? sg[0] : sg[1]);
          oth[j] = __shfl_xor(send, 16, 32);
        }

        // Low lanes own everything: r=sg[0..4], z={sg5,sg6,sg7,oth5,oth6},
        // i_n=oth[0..4], h_n=d2[0..4].
        float nn[5];
        #pragma unroll
        for (int j = 0; j < 5; ++j) nn[j] = tanh_fast(oth[j] + sg[j] * d2[j]);
        float z[5] = { sg[5], sg[6], sg[7], oth[5], oth[6] };
        #pragma unroll
        for (int j = 0; j < 5; ++j) h[d][j] = (1.0f - z[j]) * nn[j] + z[j] * h[d][j];

        if (!hi) {
          #pragma unroll
          for (int j = 0; j < 5; ++j) lds[wo[d] + j] = (_Float16)h[d][j];
        }
        ro[d] += rs[d];
        wo[d] += ws[d];
      }
    }
    int tmp = inBase; inBase = outBase; outBase = tmp;
  }

  // Final: layer-4 output at t = T-1, k = 0..9  ->  out [B,1,10] fp32
  for (int idx = lane; idx < 16 * 10; idx += 32) {
    int b = idx / 10, k = idx % 10;
    out[(gbat0 + b) * 10 + k] = (float)lds[inBase + ((T_LEN - 1) * 16 + b) * REC + k];
  }
}

extern "C" void kernel_launch(void* const* d_in, const int* in_sizes, int n_in,
                              void* d_out, int out_size, void* d_ws, size_t ws_size,
                              hipStream_t stream) {
  const float* x     = (const float*)d_in[0];
  const float* w_ih0 = (const float*)d_in[1];
  const float* w_hh0 = (const float*)d_in[2];
  const float* b_ih0 = (const float*)d_in[3];
  const float* b_hh0 = (const float*)d_in[4];
  const float* w_ihL = (const float*)d_in[5];
  const float* w_hhL = (const float*)d_in[6];
  const float* b_ihL = (const float*)d_in[7];
  const float* b_hhL = (const float*)d_in[8];
  float* out = (float*)d_out;

  const int B = in_sizes[0] / T_LEN;       // 262144
  const int batchPerBlock = WAVES * 16;    // 32
  dim3 grid((B + batchPerBlock - 1) / batchPerBlock), block(WAVES * 32);
  gru_bidir_fused<<<grid, block, 0, stream>>>(x, w_ih0, w_hh0, b_ih0, b_hh0,
                                              w_ihL, w_hhL, b_ihL, b_hhL, out);
}
